// EarlyFusionGRU_40407052320942
// MI455X (gfx1250) — matrix operand order repaired and checked by reading, not compiled, not run on hardware
//
#include <hip/hip_runtime.h>
#include <hip/hip_bf16.h>

typedef __attribute__((ext_vector_type(16))) __bf16 v16bf;
typedef __attribute__((ext_vector_type(8)))  __bf16 v8bf;
typedef __attribute__((ext_vector_type(8)))  float  v8f;

__device__ __forceinline__ v8f wmma_bf16(v16bf a, v16bf b, v8f c) {
  // D = A(16x32 bf16) x B(32x16 bf16) + C(16x16 f32)
  return __builtin_amdgcn_wmma_f32_16x16x32_bf16(false, a, false, b, (short)0, c, false, false);
}

// Workgroup barrier that orders ONLY LDS traffic (s_wait_dscnt), leaving
// global loads (LOADcnt) outstanding so register-staged prefetches can stay
// in flight across the barrier. CDNA5 split counters make this legal:
// LDS ordering needs DScnt only; private VGPR loads need no cross-thread fence.
__device__ __forceinline__ void lds_barrier() {
  asm volatile(
      "s_wait_dscnt 0\n\t"
      "s_barrier_signal -1\n\t"
      "s_barrier_wait -1"
      ::: "memory");
}

__device__ __forceinline__ v8f splat8(float v) {
  v8f r;
#pragma unroll
  for (int i = 0; i < 8; ++i) r[i] = v;
  return r;
}

// A fragment: 16-row x 32-col bf16 tile, row-major in LDS with stride ld.
// Lane L: row = L%16, khalf = L/16. f[0..7] = k(kbase+8*kh ..+7), f[8..15] = +16.
__device__ __forceinline__ v16bf load_afrag(const __bf16* S, int ld, int kbase) {
  int lane = threadIdx.x & 31;
  const __bf16* p = S + (lane & 15) * ld + kbase + ((lane >> 4) & 1) * 8;
  v16bf f;
#pragma unroll
  for (int i = 0; i < 8; ++i) { f[i] = p[i]; f[8 + i] = p[16 + i]; }
  return f;
}

// B fragment = (W^T) chunk for D = A x B with B[k][n] = W[n][k].
// Dense bf16 B 32x16, same k order as A: lane L, kh=L/16: f[0..7] = K 8*kh..+7, f[8..15] = +16.
// => lane L holds row n = nbase+L%16 of W at k = kbase + 8*kh + {0..7, 16..23}.
// Rows >= nmax or k >= kmax read as zero (padding).
__device__ __forceinline__ v16bf load_bfrag(const float* W, int nbase, int nmax,
                                            int ldw, int kbase, int kmax) {
  int lane = threadIdx.x & 31;
  int n = nbase + (lane & 15);
  int kb = kbase + ((lane >> 4) & 1) * 8;
  v16bf f;
#pragma unroll
  for (int i = 0; i < 16; ++i) {
    int k = kb + (i < 8 ? i : 8 + i);
    float w = (n < nmax && k < kmax) ? W[(size_t)n * ldw + k] : 0.0f;
    f[i] = (__bf16)w;
  }
  return f;
}

__device__ __forceinline__ float fast_sigmoid(float x) { return 1.0f / (1.0f + __expf(-x)); }
__device__ __forceinline__ float fast_tanh(float x)    { return 1.0f - 2.0f / (__expf(2.0f * x) + 1.0f); }

struct GruDir {
  const float* x;    // (256, T, I)
  const float* Wi;   // (384, I)
  const float* Wh;   // (384, 128)
  const float* bi;   // (384)
  const float* bh;   // (384)
  int I, t0, t_step, n_steps, out_col;
};
struct GruDir4 { GruDir d[4]; };

// One workgroup = 16 batch rows of one direction. 256 threads = 8 waves.
// Wave w owns N-tiles {w, w+8, w+16} of the 16x384 gate pre-activations:
//   tile w     -> r-gate cols [16w,16w+16)
//   tile w+8   -> z-gate cols
//   tile w+16  -> n-gate cols (separate xn / hn accumulators)
// x_t tiles are software-pipelined through registers; the LDS-only barrier
// lets the prefetch's s_wait_loadcnt sink to the NEXT iteration's commit,
// hiding HBM latency behind a full step of GEMM + gate math.
__global__ __launch_bounds__(256) void gru_scan_kernel(GruDir4 all, int T,
                                                       float* __restrict__ h_out) {
  const GruDir g = all.d[blockIdx.y];
  const int b0row = blockIdx.x * 16;
  const int I = g.I;

  __shared__ __bf16 sh_x[16 * 64];    // x_t tile, zero-padded to K=64
  __shared__ __bf16 sh_h[16 * 128];   // h state (bf16, WMMA A operand)
  __shared__ float  sh_hf[16 * 128];  // h state (f32 master copy)
  __shared__ float  sh_Sr[16 * 128];  // r-gate pre-activation (x+h paths + biases)
  __shared__ float  sh_Sz[16 * 128];  // z-gate pre-activation
  __shared__ float  sh_xn[16 * 128];  // n-gate x-path (+bi)
  __shared__ float  sh_hn[16 * 128];  // n-gate h-path (+bh)

  const int wave = threadIdx.x >> 5;
  const int lane = threadIdx.x & 31;
  const int nc   = lane & 15;
  const int kh   = (lane >> 4) & 1;
  const int nb0 = wave * 16;          // r tile
  const int nb1 = 128 + wave * 16;    // z tile
  const int nb2 = 256 + wave * 16;    // n tile

  // ---- preload loop-invariant weight fragments into VGPRs (bf16) ----
  v16bf wi0[2], wi1[2], wi2[2];
#pragma unroll
  for (int kc = 0; kc < 2; ++kc) {
    wi0[kc] = load_bfrag(g.Wi, nb0, 384, I, kc * 32, I);
    wi1[kc] = load_bfrag(g.Wi, nb1, 384, I, kc * 32, I);
    wi2[kc] = load_bfrag(g.Wi, nb2, 384, I, kc * 32, I);
  }
  v16bf wh0[4], wh1[4], wh2[4];
#pragma unroll
  for (int kc = 0; kc < 4; ++kc) {
    wh0[kc] = load_bfrag(g.Wh, nb0, 384, 128, kc * 32, 128);
    wh1[kc] = load_bfrag(g.Wh, nb1, 384, 128, kc * 32, 128);
    wh2[kc] = load_bfrag(g.Wh, nb2, 384, 128, kc * 32, 128);
  }
  const float bias_r  = g.bi[nb0 + nc] + g.bh[nb0 + nc];
  const float bias_z  = g.bi[nb1 + nc] + g.bh[nb1 + nc];
  const float bias_xn = g.bi[nb2 + nc];
  const float bias_hn = g.bh[nb2 + nc];

  // ---- h = 0 ----
  for (int idx = threadIdx.x; idx < 16 * 128; idx += 256) {
    sh_hf[idx] = 0.0f;
    sh_h[idx]  = (__bf16)0.0f;
  }

  // ---- register-staged x pipeline: each thread owns 4 elements of 16x64 tile
  int xm[4], xk[4];
  float xreg[4];
#pragma unroll
  for (int e = 0; e < 4; ++e) {
    const int idx = threadIdx.x + 256 * e;
    xm[e] = idx >> 6;
    xk[e] = idx & 63;
  }
#pragma unroll
  for (int e = 0; e < 4; ++e) {
    xreg[e] = (xk[e] < I)
        ? g.x[((size_t)(b0row + xm[e]) * T + g.t0) * I + xk[e]]
        : 0.0f;
  }

  for (int s = 0; s < g.n_steps; ++s) {
    // phase A: commit staged x_t to LDS (bf16), then prefetch x_{t+1}.
#pragma unroll
    for (int e = 0; e < 4; ++e)
      sh_x[threadIdx.x + 256 * e] = (__bf16)xreg[e];
    if (s + 1 < g.n_steps) {
      const int tn = g.t0 + (s + 1) * g.t_step;
#pragma unroll
      for (int e = 0; e < 4; ++e)
        xreg[e] = (xk[e] < I)
            ? g.x[((size_t)(b0row + xm[e]) * T + tn) * I + xk[e]]
            : 0.0f;
    }
    lds_barrier();  // x + h ready (DScnt only: prefetch stays in flight)

    // phase B: gate GEMM  (16x384) = x_t(16x64) Wi^T + h(16x128) Wh^T + biases
    v8f acc_r  = splat8(bias_r);
    v8f acc_z  = splat8(bias_z);
    v8f acc_xn = splat8(bias_xn);
    v8f acc_hn = splat8(bias_hn);
#pragma unroll
    for (int kc = 0; kc < 2; ++kc) {
      v16bf af = load_afrag(sh_x, 64, kc * 32);
      acc_r  = wmma_bf16(af, wi0[kc], acc_r);
      acc_z  = wmma_bf16(af, wi1[kc], acc_z);
      acc_xn = wmma_bf16(af, wi2[kc], acc_xn);
    }
#pragma unroll
    for (int kc = 0; kc < 4; ++kc) {
      v16bf af = load_afrag(sh_h, 128, kc * 32);
      acc_r  = wmma_bf16(af, wh0[kc], acc_r);
      acc_z  = wmma_bf16(af, wh1[kc], acc_z);
      acc_hn = wmma_bf16(af, wh2[kc], acc_hn);
    }
    // scatter accumulators to gate buffers (C/D layout: reg r -> M = r + 8*kh, N = nc)
    {
      const int col = wave * 16 + nc;
#pragma unroll
      for (int r = 0; r < 8; ++r) {
        const int m = r + kh * 8;
        sh_Sr[m * 128 + col] = acc_r[r];
        sh_Sz[m * 128 + col] = acc_z[r];
        sh_xn[m * 128 + col] = acc_xn[r];
        sh_hn[m * 128 + col] = acc_hn[r];
      }
    }
    lds_barrier();  // gate pre-activations ready (DScnt only)

    // phase C: elementwise gate math, update h (each thread owns 8 elements)
    {
      const int m  = threadIdx.x >> 4;
      const int c0 = (threadIdx.x & 15) * 8;
#pragma unroll
      for (int j = 0; j < 8; ++j) {
        const int c = c0 + j;
        const float r  = fast_sigmoid(sh_Sr[m * 128 + c]);
        const float z  = fast_sigmoid(sh_Sz[m * 128 + c]);
        const float n  = fast_tanh(sh_xn[m * 128 + c] + r * sh_hn[m * 128 + c]);
        const float hp = sh_hf[m * 128 + c];
        const float hn_ = (1.0f - z) * n + z * hp;
        sh_hf[m * 128 + c] = hn_;
        sh_h[m * 128 + c]  = (__bf16)hn_;
      }
    }
    // next iteration's post-phase-A barrier orders C -> B
  }

  __syncthreads();
  // epilogue: write final h tile into the fused feature matrix (256 x 512)
  for (int idx = threadIdx.x; idx < 16 * 128; idx += 256) {
    const int m = idx >> 7, c = idx & 127;
    h_out[(size_t)(b0row + m) * 512 + g.out_col + c] = sh_hf[idx];
  }
}

// Fused MLP + heads: h(256,512) -> relu(fc1 512->256) -> relu(fc2 256->128)
// -> lm = h2 wm^T + bm (20), la = h2 wa^T + ba (30).
// One workgroup per 16 batch rows, 8 waves.
__global__ __launch_bounds__(256) void fuse_head_kernel(
    const float* __restrict__ h_all,
    const float* __restrict__ w1, const float* __restrict__ b1,
    const float* __restrict__ w2, const float* __restrict__ b2,
    const float* __restrict__ wm, const float* __restrict__ bm,
    const float* __restrict__ wa, const float* __restrict__ ba,
    float* __restrict__ out) {
  const int b0row = blockIdx.x * 16;
  __shared__ __bf16 sh_in[16 * 512];
  __shared__ __bf16 sh_h1[16 * 256];
  __shared__ __bf16 sh_h2[16 * 128];

  const int wave = threadIdx.x >> 5;
  const int lane = threadIdx.x & 31;
  const int nc   = lane & 15;
  const int kh   = (lane >> 4) & 1;

  // stage input tile as bf16
  for (int idx = threadIdx.x; idx < 16 * 512; idx += 256) {
    const int m = idx >> 9, c = idx & 511;
    sh_in[idx] = (__bf16)h_all[(size_t)(b0row + m) * 512 + c];
  }
  __syncthreads();

  // fc1: N=256 (16 tiles, 2 per wave), K=512
  {
    const int nbA = wave * 16, nbB = 128 + wave * 16;
    v8f aA = splat8(b1[nbA + nc]);
    v8f aB = splat8(b1[nbB + nc]);
#pragma unroll
    for (int kc = 0; kc < 16; ++kc) {
      v16bf af = load_afrag(sh_in, 512, kc * 32);
      aA = wmma_bf16(af, load_bfrag(w1, nbA, 256, 512, kc * 32, 512), aA);
      aB = wmma_bf16(af, load_bfrag(w1, nbB, 256, 512, kc * 32, 512), aB);
    }
#pragma unroll
    for (int r = 0; r < 8; ++r) {
      const int m = r + kh * 8;
      sh_h1[m * 256 + nbA + nc] = (__bf16)fmaxf(aA[r], 0.0f);
      sh_h1[m * 256 + nbB + nc] = (__bf16)fmaxf(aB[r], 0.0f);
    }
  }
  __syncthreads();

  // fc2: N=128 (8 tiles, 1 per wave), K=256
  {
    const int nb = wave * 16;
    v8f a = splat8(b2[nb + nc]);
#pragma unroll
    for (int kc = 0; kc < 8; ++kc) {
      v16bf af = load_afrag(sh_h1, 256, kc * 32);
      a = wmma_bf16(af, load_bfrag(w2, nb, 128, 256, kc * 32, 256), a);
    }
#pragma unroll
    for (int r = 0; r < 8; ++r) {
      const int m = r + kh * 8;
      sh_h2[m * 128 + nb + nc] = (__bf16)fmaxf(a[r], 0.0f);
    }
  }
  __syncthreads();

  // heads: waves 0,1 -> lm (N=20, 2 padded tiles); waves 2,3 -> la (N=30)
  if (wave < 4) {
    const float* W  = (wave < 2) ? wm : wa;
    const float* bb = (wave < 2) ? bm : ba;
    const int nmax  = (wave < 2) ? 20 : 30;
    const int nb    = (wave & 1) * 16;
    const int n     = nb + nc;
    v8f a = splat8((n < nmax) ? bb[n] : 0.0f);
#pragma unroll
    for (int kc = 0; kc < 4; ++kc) {
      v16bf af = load_afrag(sh_h2, 128, kc * 32);
      a = wmma_bf16(af, load_bfrag(W, nb, nmax, 128, kc * 32, 128), a);
    }
    if (n < nmax) {
      const size_t base = (wave < 2) ? 0 : (size_t)256 * 20;
#pragma unroll
      for (int r = 0; r < 8; ++r) {
        const int m = r + kh * 8;
        out[base + (size_t)(b0row + m) * nmax + n] = a[r];
      }
    }
  }
}

extern "C" void kernel_launch(void* const* d_in, const int* in_sizes, int n_in,
                              void* d_out, int out_size, void* d_ws, size_t ws_size,
                              hipStream_t stream) {
  const float* g_seq = (const float*)d_in[0];
  const float* a_seq = (const float*)d_in[1];
  const float* g_wif = (const float*)d_in[2];
  const float* g_whf = (const float*)d_in[3];
  const float* g_bif = (const float*)d_in[4];
  const float* g_bhf = (const float*)d_in[5];
  const float* g_wib = (const float*)d_in[6];
  const float* g_whb = (const float*)d_in[7];
  const float* g_bib = (const float*)d_in[8];
  const float* g_bhb = (const float*)d_in[9];
  const float* a_wif = (const float*)d_in[10];
  const float* a_whf = (const float*)d_in[11];
  const float* a_bif = (const float*)d_in[12];
  const float* a_bhf = (const float*)d_in[13];
  const float* a_wib = (const float*)d_in[14];
  const float* a_whb = (const float*)d_in[15];
  const float* a_bib = (const float*)d_in[16];
  const float* a_bhb = (const float*)d_in[17];
  const float* fuse_w1 = (const float*)d_in[18];
  const float* fuse_b1 = (const float*)d_in[19];
  const float* fuse_w2 = (const float*)d_in[20];
  const float* fuse_b2 = (const float*)d_in[21];
  const float* wm = (const float*)d_in[22];
  const float* bm = (const float*)d_in[23];
  const float* wa = (const float*)d_in[24];
  const float* ba = (const float*)d_in[25];

  float* h_all = (float*)d_ws;  // (256, 512) fused features

  const int T = 512;
  GruDir4 cfg;
  // forward scans: full 512 steps; backward "last output" = exactly ONE step at t=T-1
  cfg.d[0] = { g_seq, g_wif, g_whf, g_bif, g_bhf, 63, 0,     1, 512, 0   };
  cfg.d[1] = { g_seq, g_wib, g_whb, g_bib, g_bhb, 63, T - 1, -1, 1,  128 };
  cfg.d[2] = { a_seq, a_wif, a_whf, a_bif, a_bhf, 64, 0,     1, 512, 256 };
  cfg.d[3] = { a_seq, a_wib, a_whb, a_bib, a_bhb, 64, T - 1, -1, 1,  384 };

  gru_scan_kernel<<<dim3(16, 4), dim3(256), 0, stream>>>(cfg, T, h_all);
  fuse_head_kernel<<<dim3(16), dim3(256), 0, stream>>>(
      h_all, fuse_w1, fuse_b1, fuse_w2, fuse_b2, wm, bm, wa, ba, (float*)d_out);
}
